// DSTABlock_970662609355
// MI455X (gfx1250) — compile-verified, matrix operand order checked by reading, not run on hardware
//
#include <hip/hip_runtime.h>
#include <hip/hip_bf16.h>
#include <math.h>
#include <stdint.h>

// ---------------- CDNA5 WMMA plumbing (wave32, bf16 -> f32 accum) ----------
typedef __attribute__((ext_vector_type(16))) __bf16 v16bf;
typedef __attribute__((ext_vector_type(8)))  float  v8f;
typedef __attribute__((ext_vector_type(4)))  int    v4i;

#define AS1 __attribute__((address_space(1)))
#define AS3 __attribute__((address_space(3)))

#if defined(__has_builtin)
#if __has_builtin(__builtin_amdgcn_global_load_async_to_lds_b128) && \
    __has_builtin(__builtin_amdgcn_s_wait_asynccnt)
#define HAVE_ASYNC_LDS 1
#endif
#endif
#ifndef HAVE_ASYNC_LDS
#define HAVE_ASYNC_LDS 0
#endif

__device__ __forceinline__ v8f wmma_bf16(v16bf a, v16bf b, v8f c) {
    // emits v_wmma_f32_16x16x32_bf16
    return __builtin_amdgcn_wmma_f32_16x16x32_bf16(
        /*neg_a=*/false, a, /*neg_b=*/false, b,
        /*c_mod=*/(short)0, c, /*reuse_a=*/false, /*reuse_b=*/false);
}

// 16-bit A/B fragment load from LDS, row-major [row][ld] with k contiguous.
// ISA layout (16-bit A 16x32): lane L: m=L%16, kbase=(L<16)?0:8,
// elem e<8 -> k=kbase+e ; e>=8 -> k=kbase+16+(e-8).
__device__ __forceinline__ v16bf load_frag16(const __bf16* base, int ld,
                                             int row0, int koff = 0) {
    int lane = threadIdx.x & 31;
    const __bf16* p = base + (size_t)(row0 + (lane & 15)) * ld + koff
                      + ((lane & 16) ? 8 : 0);
    v16bf f;
#pragma unroll
    for (int e = 0; e < 8; ++e)  f[e] = p[e];
#pragma unroll
    for (int e = 8; e < 16; ++e) f[e] = p[e + 8];
    return f;
}

__device__ __forceinline__ float toF(float x) { return x; }
__device__ __forceinline__ float toF(__bf16 x) { return (float)x; }

#define TT   256
#define VV   48
#define TVN  (TT * VV)       // 12288
#define CIN  256

// ---------------------------------------------------------------------------
// Generic WMMA "conv as GEMM" kernel, double-buffered through LDS.
//   Out[b, m, n] = bias[m] + sum_dt sum_c A[m, c, dt] * In[b, c, n + (dt-KW/2)*V]
// A element addr: m*a_ostride + c*a_kstride + dt   (covers (O,C) and (O,C,KW,1))
// Also accumulates GroupNorm sum/sumsq per (b, m/group_ch) into stats.
// Block: 256 thr (8 waves). Tile BM=128, BN=64, BK=32. Wave grid 4x2, each
// wave owns a 32x32 region (2x2 WMMA tiles).
// ---------------------------------------------------------------------------
template <typename TIN>
__global__ void gemm_wmma_kernel(const float* __restrict__ A,
                                 int a_ostride, int a_kstride,
                                 const TIN* __restrict__ In,
                                 const float* __restrict__ bias,
                                 __bf16* __restrict__ Out,
                                 float* __restrict__ stats,  // may be null
                                 int OC, int KW, int group_ch) {
    const int b  = blockIdx.z;
    const int n0 = blockIdx.x * 64;
    const int m0 = blockIdx.y * 128;
    const TIN* Bb = In + (size_t)b * CIN * TVN;

    __shared__ __bf16 As[2][128][40];
    __shared__ __bf16 Bs[2][64][40];
    __shared__ float  redS[256];
    __shared__ float  redQ[256];

    const int tid  = threadIdx.x;
    const int lane = tid & 31;
    const int wv   = tid >> 5;
    const int mw   = wv >> 1;   // 0..3 -> 32-row band
    const int nw   = wv & 1;    // 0..1 -> 32-col band

    v8f acc[2][2] = {};

    const int half   = KW >> 1;
    const int nsteps = KW * (CIN / 32);

    float aReg[16];
    float bReg[8];

    auto loadStep = [&](int step) {
        int dt = step >> 3;          // 8 k-chunks per tap
        int kc = (step & 7) * 32;
        int shift = (dt - half) * VV;
#pragma unroll
        for (int it = 0; it < 16; ++it) {
            int i = tid + 256 * it;
            int am = i >> 5, ak = i & 31;
            aReg[it] = A[(size_t)(m0 + am) * a_ostride
                         + (size_t)(kc + ak) * a_kstride + dt];
        }
#pragma unroll
        for (int it = 0; it < 8; ++it) {
            int i = tid + 256 * it;
            int kr = i >> 6, nn = i & 63;
            int ng = n0 + nn + shift;
            bReg[it] = ((unsigned)ng < (unsigned)TVN)
                           ? toF(Bb[(size_t)(kc + kr) * TVN + ng]) : 0.f;
        }
    };
    auto storeStep = [&](int buf) {
#pragma unroll
        for (int it = 0; it < 16; ++it) {
            int i = tid + 256 * it;
            As[buf][i >> 5][i & 31] = (__bf16)aReg[it];
        }
#pragma unroll
        for (int it = 0; it < 8; ++it) {
            int i = tid + 256 * it;
            Bs[buf][i & 63][i >> 6] = (__bf16)bReg[it];   // transposed: [n][k]
        }
    };

    loadStep(0);
    storeStep(0);
    for (int s = 0; s < nsteps; ++s) {
        __syncthreads();
        if (s + 1 < nsteps) loadStep(s + 1);   // global loads in flight
        const int cur = s & 1;
        v16bf a0 = load_frag16(&As[cur][0][0], 40, mw * 32 + 0);
        v16bf a1 = load_frag16(&As[cur][0][0], 40, mw * 32 + 16);
        v16bf b0 = load_frag16(&Bs[cur][0][0], 40, nw * 32 + 0);
        v16bf b1 = load_frag16(&Bs[cur][0][0], 40, nw * 32 + 16);
        acc[0][0] = wmma_bf16(a0, b0, acc[0][0]);
        acc[0][1] = wmma_bf16(a0, b1, acc[0][1]);
        acc[1][0] = wmma_bf16(a1, b0, acc[1][0]);
        acc[1][1] = wmma_bf16(a1, b1, acc[1][1]);
        if (s + 1 < nsteps) storeStep((s + 1) & 1);
    }

    // epilogue: bias, store bf16, GN partial stats
    float lsum = 0.f, lsq = 0.f;
    __bf16* Ob = Out + (size_t)b * OC * TVN;
    const int mhi = (lane & 16) ? 8 : 0;
#pragma unroll
    for (int mi = 0; mi < 2; ++mi)
#pragma unroll
        for (int ni = 0; ni < 2; ++ni)
#pragma unroll
            for (int j = 0; j < 8; ++j) {
                int m = m0 + mw * 32 + mi * 16 + j + mhi;
                int n = n0 + nw * 32 + ni * 16 + (lane & 15);
                float v = acc[mi][ni][j] + bias[m];
                Ob[(size_t)m * TVN + n] = (__bf16)v;
                lsum += v; lsq += v * v;
            }
    if (stats) {
        redS[tid] = lsum; redQ[tid] = lsq;
        __syncthreads();
        if (lane == 0) {
            float s = 0.f, q = 0.f;
#pragma unroll
            for (int i = 0; i < 32; ++i) { s += redS[wv * 32 + i]; q += redQ[wv * 32 + i]; }
            int g = (m0 + mw * 32) / group_ch;  // whole wave in one group
            atomicAdd(&stats[(size_t)(b * 8 + g) * 2 + 0], s);
            atomicAdd(&stats[(size_t)(b * 8 + g) * 2 + 1], q);
        }
    }
}

// ---------------------------------------------------------------------------
// Fused attention: one wave per (b, s, t).
//  q,k normalized on load; attn = q^T k / sqrt(32) + bias ; softmax ; attn x V
// ---------------------------------------------------------------------------
__global__ void attention_kernel(const __bf16* __restrict__ qk,   // (B,512,TVN)
                                 const float* __restrict__ mrQK,  // (B,8,2)
                                 const float* __restrict__ gamma,
                                 const float* __restrict__ beta,
                                 const float* __restrict__ biasm, // (8,48,48)
                                 const __bf16* __restrict__ vvb,  // (B,256,TVN)
                                 float* __restrict__ outp) {      // (B,256,TVN)
    const int t = blockIdx.x, s = blockIdx.y, b = blockIdx.z;
    const int lane = threadIdx.x;  // 32 threads

    __shared__ __bf16 qs[48][40];   // A for score gemm: qs[v][h]
    __shared__ __bf16 ks[48][40];   // B for score gemm: ks[w][h]
    __shared__ float  at[48][48];   // scores f32
    __shared__ __bf16 vs[32][72];   // A for out gemm: vs[h][w] (K pad 64)
    __shared__ __bf16 ab[48][72];   // B for out gemm: ab[v][w] (K pad 64)

    {   // load + lazy GroupNorm of q/k rows (channel = lane per head)
        int hq = s * 32 + lane;
        int hk = 256 + s * 32 + lane;
        const float* mr = mrQK + (size_t)b * 16;
        float muQ = mr[(hq >> 6) * 2], rsQ = mr[(hq >> 6) * 2 + 1];
        float muK = mr[(hk >> 6) * 2], rsK = mr[(hk >> 6) * 2 + 1];
        float gQ = gamma[hq], bQ = beta[hq];
        float gK = gamma[hk], bK = beta[hk];
        const __bf16* qrow = qk + ((size_t)b * 512 + hq) * TVN + t * VV;
        const __bf16* krow = qk + ((size_t)b * 512 + hk) * TVN + t * VV;
#pragma unroll
        for (int v = 0; v < 48; ++v) {
            qs[v][lane] = (__bf16)(((float)qrow[v] - muQ) * rsQ * gQ + bQ);
            ks[v][lane] = (__bf16)(((float)krow[v] - muK) * rsK * gK + bK);
        }
        const __bf16* vrow = vvb + ((size_t)b * 256 + s * 32 + lane) * TVN + t * VV;
#pragma unroll
        for (int w = 0; w < 48; ++w) vs[lane][w] = vrow[w];
#pragma unroll
        for (int w = 48; w < 64; ++w) vs[lane][w] = (__bf16)0.f;
    }
    __syncthreads();

    // scores: 3x3 tiles of 16x16, K=32 exactly
    v8f sc[3][3] = {};
    v16bf qa[3], kb[3];
#pragma unroll
    for (int i = 0; i < 3; ++i) {
        qa[i] = load_frag16(&qs[0][0], 40, i * 16);
        kb[i] = load_frag16(&ks[0][0], 40, i * 16);
    }
#pragma unroll
    for (int vi = 0; vi < 3; ++vi)
#pragma unroll
        for (int wi = 0; wi < 3; ++wi)
            sc[vi][wi] = wmma_bf16(qa[vi], kb[wi], sc[vi][wi]);

    const float scale = 0.17677669529663687f;  // 1/sqrt(32)
    const int mhi = (lane & 16) ? 8 : 0;
#pragma unroll
    for (int vi = 0; vi < 3; ++vi)
#pragma unroll
        for (int wi = 0; wi < 3; ++wi)
#pragma unroll
            for (int j = 0; j < 8; ++j) {
                int v = vi * 16 + j + mhi;
                int w = wi * 16 + (lane & 15);
                at[v][w] = sc[vi][wi][j] * scale + biasm[((size_t)s * 48 + v) * 48 + w];
            }
    __syncthreads();

    // softmax over w, write bf16 (rows v, k=w contiguous, pad to 64)
    for (int r = lane; r < 48; r += 32) {
        float mx = -1e30f;
#pragma unroll
        for (int w = 0; w < 48; ++w) mx = fmaxf(mx, at[r][w]);
        float sm = 0.f;
#pragma unroll
        for (int w = 0; w < 48; ++w) { float e = __expf(at[r][w] - mx); at[r][w] = e; sm += e; }
        float inv = 1.f / sm;
#pragma unroll
        for (int w = 0; w < 48; ++w) ab[r][w] = (__bf16)(at[r][w] * inv);
#pragma unroll
        for (int w = 48; w < 64; ++w) ab[r][w] = (__bf16)0.f;
    }
    __syncthreads();

    // out[h][v] = sum_w vs[h][w] * ab[v][w] : 2x3 tiles, K=64 (2 steps)
    v8f oo[2][3] = {};
#pragma unroll
    for (int kk = 0; kk < 64; kk += 32) {
        v16bf va[2], bb[3];
#pragma unroll
        for (int hi = 0; hi < 2; ++hi) va[hi] = load_frag16(&vs[0][0], 72, hi * 16, kk);
#pragma unroll
        for (int vi = 0; vi < 3; ++vi) bb[vi] = load_frag16(&ab[0][0], 72, vi * 16, kk);
#pragma unroll
        for (int hi = 0; hi < 2; ++hi)
#pragma unroll
            for (int vi = 0; vi < 3; ++vi)
                oo[hi][vi] = wmma_bf16(va[hi], bb[vi], oo[hi][vi]);
    }
    float* ob = outp + ((size_t)b * 256 + s * 32) * TVN + t * VV;
#pragma unroll
    for (int hi = 0; hi < 2; ++hi)
#pragma unroll
        for (int vi = 0; vi < 3; ++vi)
#pragma unroll
            for (int j = 0; j < 8; ++j) {
                int h = hi * 16 + j + mhi;
                int v = vi * 16 + (lane & 15);
                ob[(size_t)h * TVN + v] = oo[hi][vi][j];
            }
}

// ------------------------- small pointwise kernels -------------------------
__global__ void zero_kernel(float* p, int n) {
    int i = blockIdx.x * blockDim.x + threadIdx.x;
    if (i < n) p[i] = 0.f;
}

__global__ void biasm_kernel(const int* __restrict__ gd,
                             const float* __restrict__ table,
                             float* __restrict__ biasm) {
    int i = blockIdx.x * 256 + threadIdx.x;  // 8*48*48
    if (i >= 8 * 48 * 48) return;
    int w = i % 48, v = (i / 48) % 48, s = i / 2304;
    int d = gd[v * 48 + w];
    d = d < 0 ? 0 : (d > 12 ? 12 : d);
    biasm[i] = table[s * 13 + d];
}

__global__ void finalize_stats_kernel(float* st, float invCnt, int n) {
    int i = blockIdx.x * blockDim.x + threadIdx.x;
    if (i >= n) return;
    float s = st[2 * i], q = st[2 * i + 1];
    float m = s * invCnt;
    float var = q * invCnt - m * m;
    st[2 * i] = m;
    st[2 * i + 1] = rsqrtf(var + 1e-5f);
}

__global__ void edge_kernel(const float* __restrict__ x,
                            const float* __restrict__ ef,   // (6,256)
                            const float* __restrict__ alpha,
                            float* __restrict__ outp) {
    const int t = blockIdx.x, b = blockIdx.y;
    const int tid = threadIdx.x;  // 256
    __shared__ float xl[256][48];
    __shared__ float efl[6][256];
    __shared__ float attl[6][48];
    const float* xb = x + (size_t)b * CIN * TVN + t * VV;
#if HAVE_ASYNC_LDS
    {   // CDNA5 async DMA: global row (192B) -> LDS row, no VGPR round-trip
        const char* g = (const char*)(xb + (size_t)tid * TVN);
        AS3 char* l = (AS3 char*)(void*)&xl[tid][0];
#pragma unroll
        for (int j = 0; j < 12; ++j)
            __builtin_amdgcn_global_load_async_to_lds_b128(
                (AS1 v4i*)(uintptr_t)(g + 16 * j), (AS3 v4i*)(l + 16 * j), 0, 0);
    }
#else
#pragma unroll
    for (int v = 0; v < 48; ++v) xl[tid][v] = xb[(size_t)tid * TVN + v];
#endif
    for (int i = tid; i < 6 * 256; i += 256) efl[i >> 8][i & 255] = ef[i];
#if HAVE_ASYNC_LDS
    __builtin_amdgcn_s_wait_asynccnt(0);
#endif
    __syncthreads();
    const float isc = 0.0625f;  // 1/sqrt(256)
    for (int d = tid; d < 6 * 48; d += 256) {
        int e = d / 48, v = d % 48;
        float sm = 0.f;
        for (int c = 0; c < 256; ++c) sm += xl[c][v] * efl[e][c];
        attl[e][v] = tanhf(sm) * isc;
    }
    __syncthreads();
    float al = alpha[0];
    float* ob = outp + (size_t)b * CIN * TVN + t * VV;
#pragma unroll
    for (int v = 0; v < 48; ++v) {
        float sm = 0.f;
#pragma unroll
        for (int e = 0; e < 6; ++e) sm += efl[e][tid] * attl[e][v];
        ob[(size_t)tid * TVN + v] += al * sm;
    }
}

__global__ void gn_relu_kernel(const __bf16* __restrict__ in,
                               const float* __restrict__ mr,
                               const float* __restrict__ g,
                               const float* __restrict__ be,
                               __bf16* __restrict__ out) {
    size_t i = (size_t)blockIdx.x * 256 + threadIdx.x;
    int c = (int)((i / TVN) & 255);
    int b = (int)(i / ((size_t)256 * TVN));
    int grp = c >> 5;
    float m = mr[(b * 8 + grp) * 2], r = mr[(b * 8 + grp) * 2 + 1];
    float v = ((float)in[i] - m) * r * g[c] + be[c];
    out[i] = (__bf16)fmaxf(v, 0.f);
}

__global__ void final_kernel(const __bf16* __restrict__ b5,
                             const __bf16* __restrict__ b7,
                             const float* __restrict__ mr5,
                             const float* __restrict__ mr7,
                             const float* __restrict__ g5, const float* __restrict__ e5,
                             const float* __restrict__ g7, const float* __restrict__ e7,
                             const float* __restrict__ x,
                             float* __restrict__ out) {
    size_t i = (size_t)blockIdx.x * 256 + threadIdx.x;
    int c = (int)((i / TVN) & 255);
    int b = (int)(i / ((size_t)256 * TVN));
    int grp = c >> 5;
    float v5 = ((float)b5[i] - mr5[(b * 8 + grp) * 2]) * mr5[(b * 8 + grp) * 2 + 1] * g5[c] + e5[c];
    float v7 = ((float)b7[i] - mr7[(b * 8 + grp) * 2]) * mr7[(b * 8 + grp) * 2 + 1] * g7[c] + e7[c];
    out[i] = fmaxf(0.5f * (v5 + v7) + x[i], 0.f);
}

// ---------------------------------------------------------------------------
extern "C" void kernel_launch(void* const* d_in, const int* in_sizes, int n_in,
                              void* d_out, int out_size, void* d_ws, size_t ws_size,
                              hipStream_t stream) {
    const float* x      = (const float*)d_in[0];
    const int*   gd     = (const int*)  d_in[1];
    const float* qkw    = (const float*)d_in[2];
    const float* qkb    = (const float*)d_in[3];
    const float* qkg    = (const float*)d_in[4];
    const float* qkbe   = (const float*)d_in[5];
    const float* vw     = (const float*)d_in[6];
    const float* vb     = (const float*)d_in[7];
    const float* btab   = (const float*)d_in[8];
    const float* ef     = (const float*)d_in[9];
    const float* alpha  = (const float*)d_in[10];
    const float* ow     = (const float*)d_in[11];
    const float* ob     = (const float*)d_in[12];
    const float* ong    = (const float*)d_in[13];
    const float* onb    = (const float*)d_in[14];
    const float* t5w    = (const float*)d_in[15];
    const float* t5b    = (const float*)d_in[16];
    const float* t5g    = (const float*)d_in[17];
    const float* t5be   = (const float*)d_in[18];
    const float* t7w    = (const float*)d_in[19];
    const float* t7b    = (const float*)d_in[20];
    const float* t7g    = (const float*)d_in[21];
    const float* t7be   = (const float*)d_in[22];
    float* out = (float*)d_out;

    char* ws = (char*)d_ws;
    size_t off = 0;
    auto alloc = [&](size_t bytes) -> void* {
        void* p = ws + off;
        off = (off + bytes + 255) & ~(size_t)255;
        return p;
    };
    const size_t NB = 16;
    __bf16* qkbuf  = (__bf16*)alloc(NB * 512 * TVN * 2);
    __bf16* vvbuf  = (__bf16*)alloc(NB * 256 * TVN * 2);
    float*  attout = (float*) alloc(NB * 256 * TVN * 4);
    __bf16* sabuf  = (__bf16*)alloc(NB * 256 * TVN * 2);
    __bf16* hbuf   = (__bf16*)alloc(NB * 256 * TVN * 2);
    __bf16* b5buf  = (__bf16*)alloc(NB * 256 * TVN * 2);
    __bf16* b7buf  = (__bf16*)alloc(NB * 256 * TVN * 2);
    float*  biasm  = (float*) alloc(8 * 48 * 48 * 4);
    float*  stQK   = (float*) alloc(16 * 8 * 2 * 4);
    float*  stSA   = (float*) alloc(16 * 8 * 2 * 4);
    float*  st5    = (float*) alloc(16 * 8 * 2 * 4);
    float*  st7    = (float*) alloc(16 * 8 * 2 * 4);

    zero_kernel<<<1, 256, 0, stream>>>(stQK, 256);
    zero_kernel<<<1, 256, 0, stream>>>(stSA, 256);
    zero_kernel<<<1, 256, 0, stream>>>(st5, 256);
    zero_kernel<<<1, 256, 0, stream>>>(st7, 256);
    biasm_kernel<<<72, 256, 0, stream>>>(gd, btab, biasm);

    // qk = qkw @ x + qkb  (OC=512, GN stats, group=64ch)
    gemm_wmma_kernel<float><<<dim3(192, 4, 16), 256, 0, stream>>>(
        qkw, 256, 1, x, qkb, qkbuf, stQK, 512, 1, 64);
    // vv = vw @ x + vb
    gemm_wmma_kernel<float><<<dim3(192, 2, 16), 256, 0, stream>>>(
        vw, 256, 1, x, vb, vvbuf, nullptr, 256, 1, 32);
    finalize_stats_kernel<<<1, 128, 0, stream>>>(stQK, 1.f / (64.f * TVN), 128);

    attention_kernel<<<dim3(256, 8, 16), 32, 0, stream>>>(
        qkbuf, stQK, qkg, qkbe, biasm, vvbuf, attout);
    edge_kernel<<<dim3(256, 16), 256, 0, stream>>>(x, ef, alpha, attout);

    // sa = ow @ attout + ob (GN stats, group=32ch)
    gemm_wmma_kernel<float><<<dim3(192, 2, 16), 256, 0, stream>>>(
        ow, 256, 1, attout, ob, sabuf, stSA, 256, 1, 32);
    finalize_stats_kernel<<<1, 128, 0, stream>>>(stSA, 1.f / (32.f * TVN), 128);
    gn_relu_kernel<<<196608, 256, 0, stream>>>(sabuf, stSA, ong, onb, hbuf);

    // temporal convs as multi-tap GEMMs
    gemm_wmma_kernel<__bf16><<<dim3(192, 2, 16), 256, 0, stream>>>(
        t5w, 256 * 5, 5, hbuf, t5b, b5buf, st5, 256, 5, 32);
    gemm_wmma_kernel<__bf16><<<dim3(192, 2, 16), 256, 0, stream>>>(
        t7w, 256 * 7, 7, hbuf, t7b, b7buf, st7, 256, 7, 32);
    finalize_stats_kernel<<<1, 128, 0, stream>>>(st5, 1.f / (32.f * TVN), 128);
    finalize_stats_kernel<<<1, 128, 0, stream>>>(st7, 1.f / (32.f * TVN), 128);

    final_kernel<<<196608, 256, 0, stream>>>(
        b5buf, b7buf, st5, st7, t5g, t5be, t7g, t7be, x, out);
}
